// CosineTopKGate_60601988547192
// MI455X (gfx1250) — compile-verified, matrix operand order checked by reading, not run on hardware
//
#include <hip/hip_runtime.h>
#include <hip/hip_bf16.h>

typedef __attribute__((ext_vector_type(16))) __bf16 v16bf;
typedef __attribute__((ext_vector_type(8)))  __bf16 v8bf;
typedef __attribute__((ext_vector_type(8)))  float  v8f;
typedef __attribute__((ext_vector_type(4)))  unsigned int v4u;
typedef __attribute__((ext_vector_type(8)))  int v8i;
typedef __attribute__((ext_vector_type(4)))  int v4i;

#define N_TOKENS   16384
#define MODEL_DIM  2048
#define PROJ_DIM   256
#define MAX_EXP    64
#define CLAMP_MAXV 4.605170185988091f   /* log(100) */
#define NEG_INF_V  -1e9f

// ---------------- pre-kernel 1: W fp32 -> bf16 (held in L2 for all blocks) ----
__global__ void cvt_w_bf16_k(const float* __restrict__ W, __bf16* __restrict__ wbf) {
  int i = blockIdx.x * blockDim.x + threadIdx.x;
  if (i < PROJ_DIM * MODEL_DIM) wbf[i] = (__bf16)W[i];
}

// ---------------- pre-kernel 2: column-normalize sim, store transposed bf16 ---
__global__ void norm_simT_k(const float* __restrict__ sim, __bf16* __restrict__ simnT) {
  __shared__ float red[256];
  int e = blockIdx.x;          // expert column 0..63
  int t = threadIdx.x;         // proj row   0..255
  float v = sim[t * MAX_EXP + e];
  red[t] = v * v;
  __syncthreads();
  for (int s = 128; s > 0; s >>= 1) {
    if (t < s) red[t] += red[t + s];
    __syncthreads();
  }
  float inv = 1.0f / fmaxf(sqrtf(red[0]), 1e-12f);
  simnT[e * PROJ_DIM + t] = (__bf16)(v * inv);   // [64][256] row-major = B-frag friendly
}

// ---------------- main fused gate kernel -------------------------------------
// block = 128 threads (4 waves); each wave: 16 tokens x 256 proj; grid = 256.
__global__ __launch_bounds__(128)
void gate_main_k(const float*  __restrict__ x,
                 const float*  __restrict__ bias,
                 const float*  __restrict__ temp,
                 const float*  __restrict__ emask,
                 const __bf16* __restrict__ wbf,
                 const __bf16* __restrict__ simnT,
                 float* __restrict__ logits_out,
                 float* __restrict__ topk_out) {
  // LDS: [0,20480)  = W K-slice, 256 rows x 40 bf16 (80B padded rows; TDM pads
  //                   16B after every 64B stored -> bank-conflict-free frags)
  //      [20480,..) = h_norm bf16, 64 rows x 264 bf16 (528B stride)
  // reused later:  ldsS = masked logits (64x64 f32), ldsP = probs (64x64 f32)
  __shared__ __align__(16) char smem[20480 + 33792];
  __bf16* ldsW = (__bf16*)smem;
  __bf16* ldsH = (__bf16*)(smem + 20480);
  float*  ldsS = (float*)smem;
  float*  ldsP = (float*)(smem + 20480);

  const int tid  = threadIdx.x;
  const int wave = tid >> 5;
  const int lane = tid & 31;
  const int col  = lane & 15;
  const int half = lane >> 4;
  const int sel  = half * 8;                // K sub-offset per ISA A/B bf16 layout

  const int tokBase = blockIdx.x * 64 + wave * 16;
  const int m       = tokBase + col;        // this lane's A-frag token row

  v8f zero8;
  #pragma unroll
  for (int i = 0; i < 8; ++i) zero8[i] = 0.0f;

  v8f acc[16];
  #pragma unroll
  for (int t = 0; t < 16; ++t) acc[t] = zero8;

  // ---- TDM descriptor (wave-uniform): 256x32 bf16 tile of W into padded LDS --
  const unsigned lds_base = (unsigned)(unsigned long long)(void*)ldsW;
  const unsigned long long wbase = (unsigned long long)(void*)wbf;

  // ---------------- GEMM1: h = x @ W^T  (K = 2048, 32 per WMMA) --------------
  for (int kb = 0; kb < MODEL_DIM / 32; ++kb) {
    if (wave == 0) {
      unsigned long long gaddr = wbase + (unsigned long long)kb * 64ull; // bytes
      v4u g0;
      g0[0] = 1u;                                        // count=1, no gather
      g0[1] = lds_base;                                  // lds_addr (bytes)
      g0[2] = (unsigned)(gaddr & 0xFFFFFFFFu);           // global_addr[31:0]
      g0[3] = (unsigned)((gaddr >> 32) & 0x1FFFFFFu)     // global_addr[56:32]
            | (2u << 30);                                // type = 2 (image)
      v8i g1;
      g1[0] = (int)((1u << 16)      // data_size = 2 bytes
                  | (1u << 20)      // pad_enable
                  | (3u << 22)      // pad_interval: 16 DWORDs (64B)
                  | (3u << 25));    // pad_amount:   4 DWORDs (16B)
      g1[1] = (int)((unsigned)(MODEL_DIM & 0xFFFF) << 16); // tensor_dim0 lo16
      g1[2] = (int)((unsigned)PROJ_DIM << 16);             // tensor_dim1 lo16
      g1[3] = (int)(32u << 16);                            // tile_dim0 = 32
      g1[4] = (int)(PROJ_DIM);                             // tile_dim1=256, tile_dim2=0
      g1[5] = (int)(MODEL_DIM);                            // tensor_dim0_stride lo32
      g1[6] = 0;
      g1[7] = 0;
      v4i g2; g2[0] = 0; g2[1] = 0; g2[2] = 0; g2[3] = 0;  // 2-D tensor: unused
      v4i g3; g3[0] = 0; g3[1] = 0; g3[2] = 0; g3[3] = 0;
      v8i g4;
      #pragma unroll
      for (int i = 0; i < 8; ++i) g4[i] = 0;               // 6-arg toolchain extra group
      __builtin_amdgcn_tensor_load_to_lds(g0, g1, g2, g3, g4, 0);
      __builtin_amdgcn_s_wait_tensorcnt(0);
    }
    __syncthreads();

    // A fragment: lane<16 holds K {0..7,16..23}; lane>=16 holds K {8..15,24..31}
    const float* xp = x + (size_t)m * MODEL_DIM + kb * 32 + sel;
    v8f lo = *(const v8f*)(xp);
    v8f hi = *(const v8f*)(xp + 16);
    if (kb + 1 < MODEL_DIM / 32) __builtin_prefetch(xp + 32, 0, 0);
    v16bf a;
    #pragma unroll
    for (int i = 0; i < 8; ++i) { a[i] = (__bf16)lo[i]; a[8 + i] = (__bf16)hi[i]; }

    #pragma unroll
    for (int t = 0; t < 16; ++t) {
      const __bf16* wr = ldsW + (t * 16 + col) * 40 + sel;
      v8bf blo = *(const v8bf*)(wr);
      v8bf bhi = *(const v8bf*)(wr + 16);
      v16bf bb;
      #pragma unroll
      for (int i = 0; i < 8; ++i) { bb[i] = blo[i]; bb[8 + i] = bhi[i]; }
      acc[t] = __builtin_amdgcn_wmma_f32_16x16x32_bf16(
          false, a, false, bb, (short)0, acc[t], false, false);
    }
    __syncthreads();
  }

  // ---------------- bias + row L2 norm ---------------------------------------
  #pragma unroll
  for (int t = 0; t < 16; ++t) {
    float bv = bias[t * 16 + col];
    #pragma unroll
    for (int r = 0; r < 8; ++r) acc[t][r] += bv;
  }

  float inv[8];
  #pragma unroll
  for (int r = 0; r < 8; ++r) {
    float s = 0.0f;
    #pragma unroll
    for (int t = 0; t < 16; ++t) { float v = acc[t][r]; s += v * v; }
    // reduce across the 16 lanes of this half (token fixed per (r,half))
    s += __shfl_xor(s, 1);
    s += __shfl_xor(s, 2);
    s += __shfl_xor(s, 4);
    s += __shfl_xor(s, 8);
    inv[r] = 1.0f / fmaxf(sqrtf(s), 1e-12f);
  }

  // store normalized h as bf16, row-major [token][k], for GEMM2 A-frag reads
  #pragma unroll
  for (int t = 0; t < 16; ++t) {
    #pragma unroll
    for (int r = 0; r < 8; ++r) {
      int row = wave * 16 + r + 8 * half;
      ldsH[row * 264 + t * 16 + col] = (__bf16)(acc[t][r] * inv[r]);
    }
  }
  __syncthreads();

  // ---------------- GEMM2: logits = h_norm @ sim_norm  (K=256) ---------------
  v8f acc2[4];
  #pragma unroll
  for (int t = 0; t < 4; ++t) acc2[t] = zero8;

  #pragma unroll
  for (int kb = 0; kb < 8; ++kb) {
    const __bf16* hp = ldsH + (wave * 16 + col) * 264 + kb * 32 + sel;
    v8bf lo = *(const v8bf*)(hp);
    v8bf hi = *(const v8bf*)(hp + 16);
    v16bf a;
    #pragma unroll
    for (int i = 0; i < 8; ++i) { a[i] = lo[i]; a[8 + i] = hi[i]; }
    #pragma unroll
    for (int t = 0; t < 4; ++t) {
      const __bf16* sp = simnT + (t * 16 + col) * PROJ_DIM + kb * 32 + sel;
      v8bf blo = *(const v8bf*)(sp);
      v8bf bhi = *(const v8bf*)(sp + 16);
      v16bf bb;
      #pragma unroll
      for (int i = 0; i < 8; ++i) { bb[i] = blo[i]; bb[8 + i] = bhi[i]; }
      acc2[t] = __builtin_amdgcn_wmma_f32_16x16x32_bf16(
          false, a, false, bb, (short)0, acc2[t], false, false);
    }
  }

  // ---------------- scale, mask, emit logits ---------------------------------
  float scale = __expf(fminf(temp[0], CLAMP_MAXV));
  #pragma unroll
  for (int t = 0; t < 4; ++t) {
    float mk = emask[t * 16 + col];
    #pragma unroll
    for (int r = 0; r < 8; ++r) {
      int row = wave * 16 + r + 8 * half;
      float lg = acc2[t][r] * scale;
      lg = (mk == 0.0f) ? NEG_INF_V : lg;
      logits_out[(size_t)(blockIdx.x * 64 + row) * MAX_EXP + t * 16 + col] = lg;
      ldsS[row * MAX_EXP + t * 16 + col] = lg;
    }
  }
  __syncthreads();

  // ---------------- softmax + threshold top-k (1 lane per token) -------------
  if (lane < 16) {
    int row  = wave * 16 + lane;
    int gtok = blockIdx.x * 64 + row;
    const float* src = ldsS + row * MAX_EXP;
    float* prow      = ldsP + row * MAX_EXP;

    float mx = -3.402823466e38f;
    for (int j = 0; j < MAX_EXP; ++j) mx = fmaxf(mx, src[j]);
    float sum = 0.0f;
    for (int j = 0; j < MAX_EXP; ++j) { float e = __expf(src[j] - mx); sum += e; prow[j] = e; }
    float rs = 1.0f / sum;
    for (int j = 0; j < MAX_EXP; ++j) prow[j] = prow[j] * rs + 1e-14f;

    // keep_i = (exclusive descending-sorted cumsum) < 1.0, stable ties by index
    int cnt = 0;
    for (int j = 0; j < MAX_EXP; ++j) {
      float sj = prow[j];
      float excl = 0.0f;
      for (int k = 0; k < MAX_EXP; ++k) {
        float sk = prow[k];
        bool before = (sk > sj) || (sk == sj && k < j);
        excl += before ? sk : 0.0f;
      }
      cnt += (excl < 1.0f) ? 1 : 0;
    }
    float am = 0.0f;
    for (int e = 0; e < MAX_EXP; ++e) am += emask[e];
    int active = (int)am;
    if (cnt > active) cnt = active;
    topk_out[gtok] = (float)cnt;
  }
}

// ---------------- host launcher ---------------------------------------------
extern "C" void kernel_launch(void* const* d_in, const int* in_sizes, int n_in,
                              void* d_out, int out_size, void* d_ws, size_t ws_size,
                              hipStream_t stream) {
  const float* x     = (const float*)d_in[0];
  const float* W     = (const float*)d_in[1];
  const float* b     = (const float*)d_in[2];
  const float* sim   = (const float*)d_in[3];
  const float* temp  = (const float*)d_in[4];
  const float* emask = (const float*)d_in[5];

  float* out_logits = (float*)d_out;                       // 16384*64
  float* out_topk   = out_logits + (size_t)N_TOKENS * MAX_EXP;

  __bf16* wbf   = (__bf16*)d_ws;                           // 256*2048 bf16 = 1 MB
  __bf16* simnT = (__bf16*)((char*)d_ws + (size_t)PROJ_DIM * MODEL_DIM * 2); // 64*256 bf16

  cvt_w_bf16_k<<<(PROJ_DIM * MODEL_DIM + 255) / 256, 256, 0, stream>>>(W, wbf);
  norm_simT_k<<<MAX_EXP, 256, 0, stream>>>(sim, simnT);
  gate_main_k<<<N_TOKENS / 64, 128, 0, stream>>>(x, b, temp, emask, wbf, simnT,
                                                 out_logits, out_topk);
}
